// SAE_19765439496370
// MI455X (gfx1250) — matrix operand order repaired and checked by reading, not compiled, not run on hardware
//
#include <hip/hip_runtime.h>
#include <stdint.h>

// ---------------- problem constants (match reference) ----------------
#define B_ROWS 8192
#define D_IN   512
#define HID    16384
#define K_TOP  32

#define M_TILE  32      // rows of x per workgroup (2 row-subtiles of 16)
#define N_CHUNK 256     // latent columns per sweep (8 waves x 32)
#define THREADS 256     // 8 wave32s

typedef __attribute__((ext_vector_type(16))) __bf16 v16bf;
typedef __attribute__((ext_vector_type(8)))  __bf16 v8bf;
typedef __attribute__((ext_vector_type(4)))  __bf16 v4bf;
typedef __attribute__((ext_vector_type(8)))  float  v8f;
typedef __attribute__((ext_vector_type(4)))  float  fvec4;

// wave32-safe lane shuffles (no wave64 assumptions)
__device__ __forceinline__ float shfl_f(float v, int srcLane) {
  return __int_as_float(__builtin_amdgcn_ds_bpermute(srcLane << 2, __float_as_int(v)));
}
__device__ __forceinline__ int shfl_i(int v, int srcLane) {
  return __builtin_amdgcn_ds_bpermute(srcLane << 2, v);
}

// ---------------- kernel 1: round W_enc (f32) to a single bf16 plane ----------------
// Precision plan: A = (x - b_dec) is split hi+lo (exact to ~16 mantissa bits),
// W_enc is rounded once to bf16. Residual error ~2^-9 relative on pre_acts.
__global__ void sae_prep_w_kernel(const float* __restrict__ W,
                                  __bf16* __restrict__ Wh) {
  const int i = (blockIdx.x * blockDim.x + threadIdx.x) << 2;   // exact coverage
  fvec4 w = __builtin_nontemporal_load((const fvec4*)(W + i));  // f32 W read once -> NT
  v4bf hv;
#pragma unroll
  for (int c = 0; c < 4; ++c) hv[c] = (__bf16)w[c];
  *(v4bf*)(Wh + i) = hv;                                        // bf16 plane: keep in L2 (RT)
}

// ---------------- kernel 2: fused encode-GEMM + streaming top-k + sparse decode ----------------
__launch_bounds__(THREADS)
__global__ void sae_fused_kernel(const float* __restrict__ x,
                                 const float* __restrict__ b_enc,
                                 const float* __restrict__ b_dec,
                                 const float* __restrict__ W_dec,
                                 const __bf16* __restrict__ Wh,
                                 float* __restrict__ out) {
  __shared__ __align__(16) __bf16 Ahs[M_TILE][D_IN];        // 32 KB (sae_in hi)
  __shared__ __align__(16) __bf16 Als[M_TILE][D_IN];        // 32 KB (sae_in lo)
  __shared__ float chunk[M_TILE][N_CHUNK + 4];              // ~33 KB relu'd pre_acts
  __shared__ float topv_s[M_TILE][K_TOP];                   // 4 KB
  __shared__ int   topi_s[M_TILE][K_TOP];                   // 4 KB

  const int tid  = threadIdx.x;
  const int wave = tid >> 5;          // 0..7
  const int lane = tid & 31;
  const int j    = lane & 15;         // lane within half-wave
  const int g    = lane >> 4;         // half-wave id (0/1)
  const int row0 = blockIdx.x * M_TILE;

  // ---- stage A: load x tile (NT: read once), subtract b_dec, split bf16 hi/lo into LDS ----
  for (int e = tid * 4; e < M_TILE * D_IN; e += THREADS * 4) {
    const int r = e >> 9;             // /512
    const int d = e & 511;
    fvec4 xv = __builtin_nontemporal_load((const fvec4*)(x + (size_t)(row0 + r) * D_IN + d));
    fvec4 bd = *(const fvec4*)(b_dec + d);
    v4bf hv, lv;
#pragma unroll
    for (int c = 0; c < 4; ++c) {
      float s = xv[c] - bd[c];
      __bf16 h = (__bf16)s;
      hv[c] = h;
      lv[c] = (__bf16)(s - (float)h);
    }
    *(v4bf*)(&Ahs[r][d]) = hv;
    *(v4bf*)(&Als[r][d]) = lv;
  }
  __syncthreads();

  // ---- per-half-wave top-k state: 2 rows per 16-lane group, 2 slots/lane/row ----
  float sv0[2] = {-1.0f, -1.0f}, sv1[2] = {-1.0f, -1.0f};   // relu outputs >= 0 > -1
  int   si0[2] = {0, 0},          si1[2] = {0, 0};
  float cmin[2] = {-1.0f, -1.0f};                            // min of 32 slots (uniform/group)
  int   cpos[2] = {0, 0};

  // WMMA fragment addressing used here (contiguous k, same order for A and B):
  //  A 16x32 bf16: lane(0-15)=row, lanes 0-15 hold K 0-15; lanes 16-31 hold K 16-31
  //  B 32x16 bf16: lane&15 = column, lanes 0-15 hold K 0-15, lanes 16-31 hold K 16-31
  const int aoff = g << 4;            // 0 or 16

#pragma unroll 1
  for (int h0 = 0; h0 < HID; h0 += N_CHUNK) {
    const int hcol0 = h0 + (wave << 5) + j;                  // column tile 0
    const __bf16* wb0 = Wh + (size_t)hcol0 * D_IN + (g << 4);
    const __bf16* wb1 = wb0 + (size_t)16 * D_IN;             // column tile 1 (+16 latents)

    // prefetch next chunk's W rows (compiler can't pipeline across the topk barrier)
    if (h0 + N_CHUNK < HID) {
      const char* p0 = (const char*)(wb0 + (size_t)N_CHUNK * D_IN);
      const char* p1 = (const char*)(wb1 + (size_t)N_CHUNK * D_IN);
#pragma unroll
      for (int off = 0; off < 1024; off += 256) {
        __builtin_prefetch(p0 + off, 0, 2);                  // -> global_prefetch_b8
        __builtin_prefetch(p1 + off, 0, 2);
      }
    }

    v8f acc00 = {0.f,0.f,0.f,0.f,0.f,0.f,0.f,0.f};
    v8f acc01 = {0.f,0.f,0.f,0.f,0.f,0.f,0.f,0.f};
    v8f acc10 = {0.f,0.f,0.f,0.f,0.f,0.f,0.f,0.f};
    v8f acc11 = {0.f,0.f,0.f,0.f,0.f,0.f,0.f,0.f};

#pragma unroll
    for (int k0 = 0; k0 < D_IN; k0 += 32) {
      v16bf b0 = *(const v16bf*)(wb0 + k0);                  // 32B contiguous (L2-resident)
      v16bf b1 = *(const v16bf*)(wb1 + k0);
      union { v16bf v; v8bf h[2]; } ah0, al0, ah1, al1;
      ah0.h[0] = *(const v8bf*)(&Ahs[j][k0 + aoff]);
      ah0.h[1] = *(const v8bf*)(&Ahs[j][k0 + aoff + 8]);
      al0.h[0] = *(const v8bf*)(&Als[j][k0 + aoff]);
      al0.h[1] = *(const v8bf*)(&Als[j][k0 + aoff + 8]);
      ah1.h[0] = *(const v8bf*)(&Ahs[16 + j][k0 + aoff]);
      ah1.h[1] = *(const v8bf*)(&Ahs[16 + j][k0 + aoff + 8]);
      al1.h[0] = *(const v8bf*)(&Als[16 + j][k0 + aoff]);
      al1.h[1] = *(const v8bf*)(&Als[16 + j][k0 + aoff + 8]);
      // 2-term bf16 compensation: (Ah + Al) * B  ~= fp32 A * bf16 W
      acc00 = __builtin_amdgcn_wmma_f32_16x16x32_bf16(false, ah0.v, false, b0, (short)0, acc00, false, false);
      acc00 = __builtin_amdgcn_wmma_f32_16x16x32_bf16(false, al0.v, false, b0, (short)0, acc00, false, false);
      acc01 = __builtin_amdgcn_wmma_f32_16x16x32_bf16(false, ah0.v, false, b1, (short)0, acc01, false, false);
      acc01 = __builtin_amdgcn_wmma_f32_16x16x32_bf16(false, al0.v, false, b1, (short)0, acc01, false, false);
      acc10 = __builtin_amdgcn_wmma_f32_16x16x32_bf16(false, ah1.v, false, b0, (short)0, acc10, false, false);
      acc10 = __builtin_amdgcn_wmma_f32_16x16x32_bf16(false, al1.v, false, b0, (short)0, acc10, false, false);
      acc11 = __builtin_amdgcn_wmma_f32_16x16x32_bf16(false, ah1.v, false, b1, (short)0, acc11, false, false);
      acc11 = __builtin_amdgcn_wmma_f32_16x16x32_bf16(false, al1.v, false, b1, (short)0, acc11, false, false);
    }

    // ---- epilogue: +b_enc, relu, stage into LDS (each acc's 8 values share one column) ----
    {
      const float be0 = b_enc[hcol0];
      const float be1 = b_enc[hcol0 + 16];
      const int c0l = (wave << 5) + j;                       // local column, tile 0
#pragma unroll
      for (int vv = 0; vv < 8; ++vv) {
        const int rA = vv + (g << 3);                        // subtile-0 rows 0..15
        const int rB = 16 + rA;                              // subtile-1 rows 16..31
        chunk[rA][c0l]      = fmaxf(acc00[vv] + be0, 0.0f);
        chunk[rA][c0l + 16] = fmaxf(acc01[vv] + be1, 0.0f);
        chunk[rB][c0l]      = fmaxf(acc10[vv] + be0, 0.0f);
        chunk[rB][c0l + 16] = fmaxf(acc11[vv] + be1, 0.0f);
      }
    }
    __syncthreads();

    // ---- streaming top-32: 16-lane group (id G) owns rows 2G and 2G+1 ----
    {
      const int G = (wave << 1) | g;
#pragma unroll
      for (int rr = 0; rr < 2; ++rr) {
        const int rown = (G << 1) | rr;
        float curmin = cmin[rr]; int curpos = cpos[rr];
        float s0 = sv0[rr], s1 = sv1[rr];
        int   i0 = si0[rr], i1 = si1[rr];
#pragma unroll 1
        for (int c0 = 0; c0 < N_CHUNK; c0 += 16) {
          float v  = chunk[rown][c0 + j];
          int   gi = h0 + c0 + j;
          bool pend = v > curmin;
          while (true) {
            unsigned bal = __builtin_amdgcn_ballot_w32(pend);
            if (bal == 0u) break;                            // uniform exit
            unsigned gm = (bal >> (g << 4)) & 0xFFFFu;       // this group's pending lanes
            if (gm) {
              const int s = __ffs((int)gm) - 1;              // deterministic pick
              const int srcLane = (g << 4) + s;
              const float cv = shfl_f(v, srcLane);
              const int   ci = shfl_i(gi, srcLane);
              if (j == s) pend = false;
              if (j == (curpos & 15)) {                      // replace current-min slot
                if (curpos & 16) { s1 = cv; i1 = ci; }
                else             { s0 = cv; i0 = ci; }
              }
              // recompute min over 32 slots (butterfly within the 16-lane group)
              float mv; int mp;
              if (s0 <= s1) { mv = s0; mp = j; } else { mv = s1; mp = j + 16; }
#pragma unroll
              for (int dlt = 1; dlt < 16; dlt <<= 1) {
                float ov = shfl_f(mv, lane ^ dlt);
                int   op = shfl_i(mp, lane ^ dlt);
                if (ov < mv || (ov == mv && op < mp)) { mv = ov; mp = op; }
              }
              curmin = mv; curpos = mp;
              pend = pend && (v > curmin);
            }
          }
        }
        cmin[rr] = curmin; cpos[rr] = curpos;
        sv0[rr] = s0; sv1[rr] = s1; si0[rr] = i0; si1[rr] = i1;
      }
    }
    __syncthreads();   // before next chunk overwrites the staging buffer
  }

  // ---- publish per-row top-k to LDS ----
  {
    const int G = (wave << 1) | g;
#pragma unroll
    for (int rr = 0; rr < 2; ++rr) {
      const int rown = (G << 1) | rr;
      topv_s[rown][j]      = sv0[rr];
      topv_s[rown][j + 16] = sv1[rr];
      topi_s[rown][j]      = si0[rr];
      topi_s[rown][j + 16] = si1[rr];
    }
  }
  __syncthreads();

  // ---- sparse decode: out[r][d] = b_dec[d] + sum_k vals * W_dec[idx][d] ----
  for (int e = tid; e < M_TILE * D_IN; e += THREADS) {
    const int r = e >> 9;
    const int d = e & 511;
    float acc = b_dec[d];
#pragma unroll 4
    for (int k = 0; k < K_TOP; ++k) {
      const float tv = topv_s[r][k];
      const int   ti = topi_s[r][k];
      acc += tv * W_dec[(size_t)ti * D_IN + d];       // 32-lane coalesced gather (L2-resident)
    }
    // output streamed once -> non-temporal store, keep L2 for W planes
    __builtin_nontemporal_store(acc, out + (size_t)(row0 + r) * D_IN + d);
  }
}

// ---------------- host launcher ----------------
extern "C" void kernel_launch(void* const* d_in, const int* in_sizes, int n_in,
                              void* d_out, int out_size, void* d_ws, size_t ws_size,
                              hipStream_t stream) {
  (void)in_sizes; (void)n_in; (void)out_size; (void)ws_size;
  const float* x     = (const float*)d_in[0];
  const float* W_enc = (const float*)d_in[1];
  const float* b_enc = (const float*)d_in[2];
  const float* W_dec = (const float*)d_in[3];
  const float* b_dec = (const float*)d_in[4];
  // d_in[5] is k (== 32), baked in as K_TOP.
  float* out = (float*)d_out;

  // workspace: single bf16 plane of W_enc (16 MB) -> stays resident in 192 MB L2
  __bf16* Wh = (__bf16*)d_ws;

  const int nW = HID * D_IN;                          // 8388608, divisible by 1024
  sae_prep_w_kernel<<<nW / (256 * 4), 256, 0, stream>>>(W_enc, Wh);
  sae_fused_kernel<<<B_ROWS / M_TILE, THREADS, 0, stream>>>(x, b_enc, b_dec, W_dec, Wh, out);
}
